// Net_59811714564360
// MI455X (gfx1250) — matrix operand order repaired and checked by reading, not compiled, not run on hardware
//
#include <hip/hip_runtime.h>
#include <math.h>

typedef _Float16 half16 __attribute__((ext_vector_type(16)));
typedef _Float16 half8  __attribute__((ext_vector_type(8)));
typedef float    f32x8  __attribute__((ext_vector_type(8)));
typedef unsigned int uint32x4 __attribute__((ext_vector_type(4)));
typedef int      int32x8 __attribute__((ext_vector_type(8)));
typedef int      int32x4 __attribute__((ext_vector_type(4)));

#define NB        4096            // batch
#define H1HW      26              // conv1 out spatial
#define C1        32
#define KFC1      9216
#define NFC1      128
#define KSTEPS_FC1 (KFC1 / 32)    // 288
#define KHALF     (KSTEPS_FC1 / 2) // 144
#define GDIM      10
#define C1_PIX    (NB * H1HW * H1HW)   // 2,768,896 (divisible by 16)
#define C1_TILES  (C1_PIX / 16)        // 173,056
#define IMG_ELT   (H1HW * H1HW * C1)   // 21,632 halfs = 43,264 B per image

// Tensor Data Mover builtin: arity differs between toolchains (bridge doc).
#if __has_include(<hip/amd_detail/amd_gfx1250_TDM.h>)
#define TDM_LOAD(g0, g1, g2, g3) \
    __builtin_amdgcn_tensor_load_to_lds(g0, g1, g2, g3, (int32x8){0,0,0,0,0,0,0,0}, 0)
#else
#define TDM_LOAD(g0, g1, g2, g3) \
    __builtin_amdgcn_tensor_load_to_lds(g0, g1, g2, g3, 0)
#endif

// ---------------------------------------------------------------------------
// Weight repack -> per-lane WMMA B-fragment layout (ISA 7.12.2):
// lane (hi = lane>>4, lm = lane&15): col N = lm, K = hi*8 + (h&7) + (h>>3)*16 (conv1 pack: K = hi*16 + h)
// ---------------------------------------------------------------------------

// conv1_w [32][1][3][3] -> padded B 32x32 f16 (K = tap 0..8, rest 0), 2 N-tiles
__global__ __launch_bounds__(256)
void pack_conv1_w(const float* __restrict__ w, _Float16* __restrict__ pk) {
    const int t = blockIdx.x * 256 + threadIdx.x;          // < 2*512
    if (t >= 2 * 512) return;
    const int h    = t & 15;
    const int lane = (t >> 4) & 31;
    const int nt   = t >> 9;
    const int n    = nt * 16 + (lane & 15);
    const int k    = (lane >> 4) * 16 + h;                 // tap index
    pk[t] = (k < 9) ? (_Float16)w[n * 9 + k] : (_Float16)0.0f;
}

// conv2_w [64][32][3][3]: K-order k = tap*32 + q, where h1 stores conv1
// channel c at interleaved position q = 2*(c%16)+(c/16)  (inverse below).
__global__ __launch_bounds__(256)
void pack_conv2_w(const float* __restrict__ w, _Float16* __restrict__ pk) {
    const int t = blockIdx.x * 256 + threadIdx.x;          // < 4*9*512
    if (t >= 4 * 9 * 512) return;
    const int h    = t & 15;
    int tmp        = t >> 4;
    const int lane = tmp & 31; tmp >>= 5;
    const int ks   = tmp % 9;
    const int nt   = tmp / 9;
    const int n    = nt * 16 + (lane & 15);
    const int q    = (lane >> 4) * 8 + (h & 7) + (h >> 3) * 16; // storage position
    const int c    = (q & 1) * 16 + (q >> 1);              // actual channel
    const int ky = ks / 3, kx = ks % 3;
    pk[t] = (_Float16)w[((size_t)(n * C1 + c) * 3 + ky) * 3 + kx];
}

// fc1_w [128][9216]
__global__ __launch_bounds__(256)
void pack_fc1_w(const float* __restrict__ w, _Float16* __restrict__ pk) {
    const int t = blockIdx.x * 256 + threadIdx.x;          // < 8*288*512
    if (t >= 8 * KSTEPS_FC1 * 512) return;
    const int h    = t & 15;
    int tmp        = t >> 4;
    const int lane = tmp & 31; tmp >>= 5;
    const int ks   = tmp % KSTEPS_FC1;
    const int nt   = tmp / KSTEPS_FC1;
    const int n    = nt * 16 + (lane & 15);
    const int k    = ks * 32 + (lane >> 4) * 8 + (h & 7) + (h >> 3) * 16;
    pk[t] = (_Float16)w[(size_t)n * KFC1 + k];
}

// ---------------------------------------------------------------------------
// conv1 as WMMA implicit GEMM (K = 9 taps zero-padded to 32, N = 32 ch).
// B fragments live in registers across a grid-stride loop. Output f16
// [pixel][32] with interleaved channel positions -> one packed b32 store/row.
// ---------------------------------------------------------------------------
__global__ __launch_bounds__(256)
void conv1_wmma(const float* __restrict__ xin, const _Float16* __restrict__ pk,
                const float* __restrict__ bias, _Float16* __restrict__ h1) {
    const int lane = threadIdx.x & 31;
    const int lm = lane & 15, hi = lane >> 4;
    const int waveId = blockIdx.x * 8 + (threadIdx.x >> 5);
    const int nWaves = gridDim.x * 8;

    const half16 b0 = *(const half16*)(pk + lane * 16);          // ch 0..15
    const half16 b1 = *(const half16*)(pk + 512 + lane * 16);    // ch 16..31
    const float bs0 = bias[lm];
    const float bs1 = bias[16 + lm];

    for (int tile = waveId; tile < C1_TILES; tile += nWaves) {
        const int p   = tile * 16 + lm;                    // this lane's A row
        const int bi  = p / (H1HW * H1HW);
        const int rem = p % (H1HW * H1HW);
        const int y = rem / H1HW, xc = rem % H1HW;
        const float* xp = xin + (size_t)(bi * 28 + y) * 28 + xc;

        float tin[9];
        #pragma unroll
        for (int t = 0; t < 9; ++t) tin[t] = xp[(t / 3) * 28 + (t % 3)];

        union { half16 v; _Float16 e[16]; } a;
        #pragma unroll
        for (int i = 0; i < 16; ++i) a.e[i] = (_Float16)0.0f;
        if (hi == 0) {                                     // lo half: k = 0..8
            #pragma unroll
            for (int t = 0; t < 9; ++t) a.e[t] = (_Float16)tin[t];
        }

        f32x8 c0 = {}, c1 = {};
        c0 = __builtin_amdgcn_wmma_f32_16x16x32_f16(false, a.v, false, b0,
                                                    (short)0, c0, false, false);
        c1 = __builtin_amdgcn_wmma_f32_16x16x32_f16(false, a.v, false, b1,
                                                    (short)0, c1, false, false);
        // one base pointer, immediate offsets, one packed b32 store per row
        _Float16* dst = h1 + ((size_t)(tile * 16) + hi * 8) * C1 + 2 * lm;
        #pragma unroll
        for (int r = 0; r < 8; ++r) {
            union { _Float16 h[2]; unsigned int u; } pk2;
            pk2.h[0] = (_Float16)fmaxf(c0[r] + bs0, 0.0f); // pos 2*lm   (ch lm)
            pk2.h[1] = (_Float16)fmaxf(c1[r] + bs1, 0.0f); // pos 2*lm+1 (ch 16+lm)
            *(unsigned int*)(dst + r * C1) = pk2.u;
        }
    }
}

// ---------------------------------------------------------------------------
// conv2 implicit GEMM + bias + ReLU + fused 2x2 maxpool.
// One block = one image. TDM stages the 43 KB conv1 slice into LDS
// (tensor_load_to_lds + s_wait_tensorcnt), then each of 4 waves owns one
// N-tile: 9 B fragments resident in registers, 36 M-tiles x 9 WMMA from LDS.
// ---------------------------------------------------------------------------
__global__ __launch_bounds__(128)
void conv2_wmma_pool(const _Float16* __restrict__ h1,
                     const _Float16* __restrict__ wpk,
                     const float* __restrict__ bias,
                     _Float16* __restrict__ pooled) {
    __shared__ _Float16 simg[IMG_ELT];                     // 43,264 B
    const int lane = threadIdx.x & 31;
    const int nt   = threadIdx.x >> 5;                     // N-tile 0..3
    const int bi   = blockIdx.x;
    const int lm = lane & 15, hi = lane >> 4;

    if (nt == 0) {                                         // wave 0 issues TDM
        const unsigned long long ga =
            (unsigned long long)(const void*)(h1 + (size_t)bi * IMG_ELT);
        const unsigned int ldsoff = (unsigned int)(size_t)(void*)&simg[0];
        const unsigned int units = (IMG_ELT * 2) / 8;      // 5408 x 8-byte units
        // D# group0: count=1 | lds_addr | global_addr | type=2 (ISA 8.3)
        uint32x4 g0 = { 1u, ldsoff, (unsigned int)ga,
                        (unsigned int)((ga >> 32) & 0x01FFFFFFu) | 0x80000000u };
        // D# group1: data_size=8B, 1-D tensor/tile of `units` (ISA 8.4)
        int32x8 g1 = { (int)(3u << 16),                    // data_size=3 (8B)
                       (int)(units << 16),                 // tensor_dim0[15:0]
                       (int)(1u << 16),                    // tensor_dim1 = 1
                       (int)(units << 16),                 // tile_dim0
                       0,                                  // tile_dim1/2 unused
                       (int)units,                         // tensor_dim0_stride
                       0, 0 };
        int32x4 gz = { 0, 0, 0, 0 };
        TDM_LOAD(g0, g1, gz, gz);
        __builtin_amdgcn_s_wait_tensorcnt(0);
    }
    __syncthreads();

    // B fragments for this wave's N-tile: resident across all 36 M-tiles
    const _Float16* bp = wpk + (size_t)nt * 9 * 512 + lane * 16;
    half16 bf[9];
    #pragma unroll
    for (int tap = 0; tap < 9; ++tap) bf[tap] = *(const half16*)(bp + tap * 512);

    const int ch = nt * 16 + lm;
    const float bv = bias[ch];
    _Float16* dstc = pooled + (size_t)bi * KFC1 + (size_t)ch * 144;

    #pragma unroll 2
    for (int mtile = 0; mtile < 36; ++mtile) {
        // A row -> pixel (quad-major: 4 consecutive m = one 2x2 pool window)
        const int m    = mtile * 16 + lm;
        const int quad = m >> 2, sub = m & 3;
        const int y = (quad / 12) * 2 + (sub >> 1);
        const int x = (quad % 12) * 2 + (sub & 1);
        const _Float16* ap = simg + (y * H1HW + x) * C1 + hi * 8;

        f32x8 acc = {};
        #pragma unroll
        for (int tap = 0; tap < 9; ++tap) {
            union { half16 v; half8 h[2]; } a;
            const int off = ((tap / 3) * H1HW + (tap % 3)) * C1;   // constant
            a.h[0] = *(const half8*)(ap + off);            // k = hi*8+0..7
            a.h[1] = *(const half8*)(ap + off + 16);       // k = 16+hi*8+0..7
            acc = __builtin_amdgcn_wmma_f32_16x16x32_f16(false, a.v, false,
                      bf[tap], (short)0, acc, false, false);
        }
        // rows hi*8+r; 4 consecutive rows = one pool quad
        const float q0 = fmaxf(fmaxf(acc[0], acc[1]), fmaxf(acc[2], acc[3]));
        const float q1 = fmaxf(fmaxf(acc[4], acc[5]), fmaxf(acc[6], acc[7]));
        const int qA = mtile * 4 + hi * 2;                 // quad = py*12+px
        dstc[qA]     = (_Float16)fmaxf(q0 + bv, 0.0f);
        dstc[qA + 1] = (_Float16)fmaxf(q1 + bv, 0.0f);
    }
}

// ---------------------------------------------------------------------------
// fc1: [4096,9216] x [9216,128], f16 WMMA, f32 acc. Each wave: one 16-row
// M-tile x 4 N-tiles x half of K -> 4 WMMA per k-step, 1024 waves.
// Partials summed (with bias+ReLU) inside fc2_gmm.
// ---------------------------------------------------------------------------
__global__ __launch_bounds__(128)
void fc1_wmma(const _Float16* __restrict__ pooled,
              const _Float16* __restrict__ wpk, float* __restrict__ part) {
    const int lane = threadIdx.x & 31;
    const int wave = threadIdx.x >> 5;
    const int ksp  = wave >> 1;                            // K split 0/1
    const int nth  = wave & 1;                             // N half 0/1
    const int mtile = blockIdx.x;                          // 0..255
    const int lm = lane & 15, hi = lane >> 4;

    const _Float16* ar = pooled + (size_t)(mtile * 16 + lm) * KFC1
                       + (size_t)ksp * KHALF * 32 + hi * 8;
    const _Float16* bb = wpk + ((size_t)nth * 4 * KSTEPS_FC1 + (size_t)ksp * KHALF) * 512
                       + lane * 16;
    f32x8 acc[4] = {};
    #pragma unroll 2
    for (int ks = 0; ks < KHALF; ++ks) {
        union { half16 v; half8 h[2]; } a;
        a.h[0] = *(const half8*)(ar + ks * 32);
        a.h[1] = *(const half8*)(ar + ks * 32 + 16);
        #pragma unroll
        for (int nt = 0; nt < 4; ++nt) {
            const half16 bfr = *(const half16*)(bb + ((size_t)nt * KSTEPS_FC1 + ks) * 512);
            acc[nt] = __builtin_amdgcn_wmma_f32_16x16x32_f16(false, a.v, false, bfr,
                                                             (short)0, acc[nt], false, false);
        }
    }
    float* pdst = part + (size_t)ksp * NB * NFC1;
    #pragma unroll
    for (int nt = 0; nt < 4; ++nt) {
        const int col = (nth * 4 + nt) * 16 + lm;
        #pragma unroll
        for (int r = 0; r < 8; ++r) {
            const int row = mtile * 16 + hi * 8 + r;
            pdst[(size_t)row * NFC1 + col] = acc[nt][r];
        }
    }
}

// ---------------------------------------------------------------------------
// GMM precompute: Sigma_inv = L^{-T} D^{-1} L^{-1}, det_scale = rsqrt(prod D)
// ---------------------------------------------------------------------------
__global__ void gmm_precompute(const float* __restrict__ raw,
                               float* __restrict__ sinv, float* __restrict__ ds) {
    const int k = threadIdx.x;
    if (k >= GDIM) return;
    const float* M = raw + k * GDIM * GDIM;
    float Dinv[GDIM], Linv[GDIM][GDIM];
    float prod = 1.0f;
    for (int a = 0; a < GDIM; ++a) {
        const float rd = M[a * GDIM + a];
        const float D = rd * rd + 1e-4f;
        Dinv[a] = 1.0f / D; prod *= D;
    }
    ds[k] = rsqrtf(prod);
    for (int j = 0; j < GDIM; ++j) {                       // forward substitution
        for (int i = 0; i < GDIM; ++i) Linv[i][j] = 0.0f;
        Linv[j][j] = 1.0f;
        for (int i = j + 1; i < GDIM; ++i) {
            float s = 0.0f;
            for (int mm = j; mm < i; ++mm) s = fmaf(M[i * GDIM + mm], Linv[mm][j], s);
            Linv[i][j] = -s;
        }
    }
    for (int i = 0; i < GDIM; ++i)
        for (int j = 0; j < GDIM; ++j) {
            float s = 0.0f;
            for (int a = 0; a < GDIM; ++a) s = fmaf(Linv[a][i] * Dinv[a], Linv[a][j], s);
            sinv[(k * GDIM + i) * GDIM + j] = s;
        }
}

// ---------------------------------------------------------------------------
// fc1 partial reduce + bias + ReLU + fc2 (128->10) + GMM posterior.
// ---------------------------------------------------------------------------
__global__ __launch_bounds__(256)
void fc2_gmm(const float* __restrict__ part, const float* __restrict__ b1,
             const float* __restrict__ w2, const float* __restrict__ b2,
             const float* __restrict__ centers, const float* __restrict__ sinv,
             const float* __restrict__ ds, float* __restrict__ out) {
    __shared__ float sw[NFC1 * GDIM];
    __shared__ float ss[GDIM * GDIM * GDIM];
    __shared__ float sc[GDIM * GDIM];
    __shared__ float s1[NFC1];
    __shared__ float sb[GDIM], sd[GDIM];
    for (int i = threadIdx.x; i < NFC1 * GDIM; i += 256) sw[i] = w2[i];
    for (int i = threadIdx.x; i < GDIM * GDIM * GDIM; i += 256) ss[i] = sinv[i];
    for (int i = threadIdx.x; i < GDIM * GDIM; i += 256) sc[i] = centers[i];
    for (int i = threadIdx.x; i < NFC1; i += 256) s1[i] = b1[i];
    if (threadIdx.x < GDIM) { sb[threadIdx.x] = b2[threadIdx.x]; sd[threadIdx.x] = ds[threadIdx.x]; }
    __syncthreads();

    const int bi = blockIdx.x * 256 + threadIdx.x;
    if (bi >= NB) return;
    const float* p0 = part + (size_t)bi * NFC1;
    const float* p1 = part + (size_t)NB * NFC1 + (size_t)bi * NFC1;
    float h2[GDIM];
    #pragma unroll
    for (int k = 0; k < GDIM; ++k) h2[k] = sb[k];
    for (int j = 0; j < NFC1; ++j) {
        const float a = fmaxf(p0[j] + p1[j] + s1[j], 0.0f);
        #pragma unroll
        for (int k = 0; k < GDIM; ++k) h2[k] = fmaf(a, sw[k * NFC1 + j], h2[k]);
    }
    float ex[GDIM], mx = -1e30f;
    #pragma unroll
    for (int k = 0; k < GDIM; ++k) {
        float diff[GDIM];
        #pragma unroll
        for (int d = 0; d < GDIM; ++d) diff[d] = h2[d] - sc[k * GDIM + d];
        float dist = 0.0f;
        #pragma unroll
        for (int i = 0; i < GDIM; ++i) {
            float row = 0.0f;
            #pragma unroll
            for (int j = 0; j < GDIM; ++j) row = fmaf(ss[(k * GDIM + i) * GDIM + j], diff[j], row);
            dist = fmaf(diff[i], row, dist);
        }
        ex[k] = -0.5f * dist;
        mx = fmaxf(mx, ex[k]);
    }
    float sum = 0.0f;
    #pragma unroll
    for (int k = 0; k < GDIM; ++k) { ex[k] = sd[k] * __expf(ex[k] - mx); sum += ex[k]; }
    const float inv = 1.0f / sum;
    #pragma unroll
    for (int k = 0; k < GDIM; ++k) out[(size_t)bi * GDIM + k] = ex[k] * inv;
}

// ---------------------------------------------------------------------------
extern "C" void kernel_launch(void* const* d_in, const int* in_sizes, int n_in,
                              void* d_out, int out_size, void* d_ws, size_t ws_size,
                              hipStream_t stream) {
    (void)in_sizes; (void)n_in; (void)out_size; (void)ws_size;
    const float* x       = (const float*)d_in[0];
    const float* c1w     = (const float*)d_in[1];
    const float* c1b     = (const float*)d_in[2];
    const float* c2w     = (const float*)d_in[3];
    const float* c2b     = (const float*)d_in[4];
    const float* f1w     = (const float*)d_in[5];
    const float* f1b     = (const float*)d_in[6];
    const float* f2w     = (const float*)d_in[7];
    const float* f2b     = (const float*)d_in[8];
    const float* centers = (const float*)d_in[9];
    const float* rawm    = (const float*)d_in[10];
    float* out = (float*)d_out;

    // workspace carve-up (256B aligned)
    char* ws = (char*)d_ws;
    size_t off = 0;
    auto take = [&](size_t bytes) { void* p = ws + off; off += (bytes + 255) & ~size_t(255); return p; };
    _Float16* h1     = (_Float16*)take((size_t)C1_PIX * C1 * 2);             // 177 MB
    _Float16* pooled = (_Float16*)take((size_t)NB * KFC1 * 2);               //  75 MB
    float*    fpart  = (float*)   take((size_t)2 * NB * NFC1 * 4);           //   4 MB
    _Float16* pkc1   = (_Float16*)take((size_t)2 * 512 * 2);
    _Float16* pkc2   = (_Float16*)take((size_t)4 * 9 * 512 * 2);
    _Float16* pkf1   = (_Float16*)take((size_t)8 * KSTEPS_FC1 * 512 * 2);
    float*    sinv   = (float*)   take((size_t)GDIM * GDIM * GDIM * 4);
    float*    dsc    = (float*)   take((size_t)GDIM * 4);

    // prep (independent)
    pack_conv1_w<<<(2 * 512 + 255) / 256, 256, 0, stream>>>(c1w, pkc1);
    pack_conv2_w<<<(4 * 9 * 512 + 255) / 256, 256, 0, stream>>>(c2w, pkc2);
    pack_fc1_w<<<(8 * KSTEPS_FC1 * 512 + 255) / 256, 256, 0, stream>>>(f1w, pkf1);
    gmm_precompute<<<1, 16, 0, stream>>>(rawm, sinv, dsc);

    // pipeline: all three GEMM stages on WMMA; conv2 staged via TDM
    conv1_wmma<<<2048, 256, 0, stream>>>(x, pkc1, c1b, h1);
    conv2_wmma_pool<<<NB, 128, 0, stream>>>(h1, pkc2, c2b, pooled);
    fc1_wmma<<<256, 128, 0, stream>>>(pooled, pkf1, fpart);
    fc2_gmm<<<(NB + 255) / 256, 256, 0, stream>>>(fpart, f1b, f2w, f2b, centers, sinv, dsc, out);
}
